// GCN_16990890622997
// MI455X (gfx1250) — matrix operand order repaired and checked by reading, not compiled, not run on hardware
//
#include <hip/hip_runtime.h>
#include <hip/hip_bf16.h>
#include <math.h>
#include <type_traits>

// ---------------------------------------------------------------------------
// GCN forward for MI455X (gfx1250, wave32).
//  - Edge scatter/gather dominates; whole working set (~80 MB) is L2-resident
//    (192 MB L2), so we optimize L2 bytes: node features stored in bf16
//    (halves gather traffic), scatter accumulation stays f32.
//  - Feature GEMMs run on v_wmma_f32_16x16x32_bf16; weights are prepacked
//    once into the exact B-fragment layout so each wave loads B with one
//    32-byte vector load instead of 16 strided b32 loads.
// ---------------------------------------------------------------------------

typedef __attribute__((ext_vector_type(16))) __bf16 v16bf;
typedef __attribute__((ext_vector_type(8)))  float  v8f;

// ---- one-time weight prepack: W[K,N] f32 -> bf16 B-fragments --------------
// Fragment order: ((tn * KT + kt) * 32 + lane) * 16 + j, with
// lane: n = tn*16 + (lane&15), k = kt*32 + 16*(lane>>4) + j  (contiguous k order, same as A in k_gemm_wmma).
template<int K, int N>
__global__ void k_pack_b(const float* __restrict__ W, __bf16* __restrict__ Bp) {
    constexpr int KT = (K + 31) / 32;
    constexpr int TN = (N + 15) / 16;
    const int t = blockIdx.x * blockDim.x + threadIdx.x;
    if (t >= TN * KT * 32) return;
    const int lane = t & 31;
    const int kt   = (t >> 5) % KT;
    const int tn   = (t >> 5) / KT;
    const int n  = tn * 16 + (lane & 15);
    const int kb = kt * 32 + 16 * (lane >> 4);
    __bf16* out = Bp + (size_t)t * 16;
    #pragma unroll
    for (int j = 0; j < 16; ++j) {
        const int k = kb + j;
        out[j] = (k < K && n < N) ? (__bf16)W[(size_t)k * N + n] : (__bf16)0.0f;
    }
}

// ---- WMMA GEMM: C[M,NS] = A[M,K] @ B[K,N]; A is f32 or bf16 (lda=LDA),
// B is prepacked bf16 fragments, C is CT with stride NS (cols [N,NS) zeroed).
template<int K, int N, int NS, int LDA, typename AT, typename CT>
__global__ void k_gemm_wmma(const AT* __restrict__ A,
                            const __bf16* __restrict__ Bp,
                            CT* __restrict__ C, int M) {
    constexpr int TN = (N + 15) / 16;
    constexpr int KT = (K + 31) / 32;
    const int lane = threadIdx.x & 31;
    const int wid  = blockIdx.x * (blockDim.x >> 5) + (threadIdx.x >> 5);
    const int tm = wid / TN, tn = wid % TN;
    const int m0 = tm * 16, n0 = tn * 16;
    if (m0 >= M) return;

    const int half = lane >> 4;      // 0 or 1
    const int l16  = lane & 15;
    const int row  = m0 + l16;       // A row this lane feeds
    const int col  = n0 + l16;       // C column this lane owns

    v8f acc = {};
    const AT* ap = A + (size_t)row * LDA;

    #pragma unroll
    for (int kt = 0; kt < KT; ++kt) {
        const int k0 = kt * 32;
        v16bf a = {};
        if (row < M) {
            #pragma unroll
            for (int j = 0; j < 8; ++j) {          // A 16x32 bf16 layout
                const int k1 = k0 + half * 16 + j; // elems 0-7: K=16h+j
                const int k2 = k1 + 8;             // elems 8-15: K=16h+8+j
                if (k1 < K) a[j]     = (__bf16)ap[k1];
                if (k2 < K) a[8 + j] = (__bf16)ap[k2];
            }
        }
        // one 32B vector load: this wave's prepacked B fragment
        const v16bf b = *(const v16bf*)(Bp + ((size_t)(tn * KT + kt) * 32 + lane) * 16);
        acc = __builtin_amdgcn_wmma_f32_16x16x32_bf16(
            false, a, false, b, (short)0, acc, false, false);
    }

    #pragma unroll
    for (int r = 0; r < 8; ++r) {                  // C/D: row = 8h+r, col = l16
        const int rr = m0 + half * 8 + r;
        if (rr < M) {
            if (col < N)       C[(size_t)rr * NS + col] = (CT)acc[r];
            else if (col < NS) C[(size_t)rr * NS + col] = (CT)0.0f;   // pad cols
        }
    }
}

// ---------------- elementwise / scatter kernels ----------------------------

__global__ void k_fill(float* __restrict__ p, float v, int n) {
    int i = blockIdx.x * blockDim.x + threadIdx.x;
    if (i < n) p[i] = v;
}

__global__ void k_deg_count(const int* __restrict__ dst, float* __restrict__ deg, int E) {
    int e = blockIdx.x * blockDim.x + threadIdx.x;
    if (e < E) atomicAdd(&deg[dst[e]], 1.0f);
}

__global__ void k_rsqrt_inplace(float* __restrict__ d, int n) {
    int i = blockIdx.x * blockDim.x + threadIdx.x;
    if (i < n) d[i] = rsqrtf(d[i]);
}

// self-loop term doubles as deterministic accumulator init:
// agg[i,f] = dis[i]^2 * h[i,f] (f < Fv), 0 on pad columns.
__global__ void k_selfloop_init(const float* __restrict__ dis, const __bf16* __restrict__ h,
                                float* __restrict__ agg, int Nn, int Fv, int S) {
    int gid = blockIdx.x * blockDim.x + threadIdx.x;
    if (gid >= Nn * S) return;
    int i = gid / S, f = gid % S;
    float w = dis[i] * dis[i];
    agg[gid] = (f < Fv) ? w * (float)h[gid] : 0.0f;
}

// one thread per (edge, 8-feature chunk): 16B bf16 gather, f32 atomic scatter.
// bf16 -> f32 is a 16-bit shift (no cvt needed).
__global__ void k_edge_agg(const int* __restrict__ src, const int* __restrict__ dst,
                           const float* __restrict__ dis, const __bf16* __restrict__ h,
                           float* __restrict__ agg, int E, int C, int S) {
    long long gid = (long long)blockIdx.x * blockDim.x + threadIdx.x;
    if (gid >= (long long)E * C) return;
    const int e = (int)(gid / C);
    const int q = (int)(gid % C);
    const int s = src[e], d = dst[e];
    const float w = dis[s] * dis[d];
    const uint4 u = *(const uint4*)(h + (size_t)s * S + q * 8);
    float* o = agg + (size_t)d * S + q * 8;
    const unsigned uu[4] = {u.x, u.y, u.z, u.w};
    #pragma unroll
    for (int i = 0; i < 4; ++i) {
        atomicAdd(o + 2 * i,     w * __uint_as_float(uu[i] << 16));
        atomicAdd(o + 2 * i + 1, w * __uint_as_float(uu[i] & 0xffff0000u));
    }
}

// bias + ReLU, emit bf16 copy (feeds next WMMA GEMM); pad columns -> 0
__global__ void k_bias_relu_bf16(const float* __restrict__ agg, const float* __restrict__ b,
                                 __bf16* __restrict__ hr, int Nn, int Fv, int S) {
    int gid = blockIdx.x * blockDim.x + threadIdx.x;
    if (gid >= Nn * S) return;
    int f = gid % S;
    hr[gid] = (f < Fv) ? (__bf16)fmaxf(agg[gid] + b[f], 0.0f) : (__bf16)0.0f;
}

// bias + ReLU in place (f32, final feature layer)
__global__ void k_bias_relu_f32(float* __restrict__ agg, const float* __restrict__ b,
                                int Nn, int Fv, int S) {
    int gid = blockIdx.x * blockDim.x + threadIdx.x;
    if (gid >= Nn * S) return;
    int f = gid % S;
    agg[gid] = (f < Fv) ? fmaxf(agg[gid] + b[f], 0.0f) : 0.0f;
}

// fused tail weights: w34[j] = (W3 @ W4)[j], w34[34] = b3.W4 + b4
__global__ void k_w34(const float* __restrict__ W3, const float* __restrict__ b3,
                      const float* __restrict__ W4, const float* __restrict__ b4,
                      float* __restrict__ w34) {
    int j = blockIdx.x * blockDim.x + threadIdx.x;
    if (j < 34) {
        float s = 0.0f;
        for (int k = 0; k < 3; ++k) s += W3[j * 3 + k] * W4[k];
        w34[j] = s;
    } else if (j == 34) {
        float c = b4[0];
        for (int k = 0; k < 3; ++k) c += b3[k] * W4[k];
        w34[34] = c;
    }
}

// per-node y = h.w34 + c, pooled per graph via atomics
__global__ void k_node_pool(const float* __restrict__ h, const float* __restrict__ w34,
                            const int* __restrict__ batch, float* __restrict__ pool,
                            int Nn, int Fv, int S) {
    int i = blockIdx.x * blockDim.x + threadIdx.x;
    if (i >= Nn) return;
    const float* hp = h + (size_t)i * S;
    float y = w34[34];
    #pragma unroll 2
    for (int f = 0; f < Fv; ++f) y += hp[f] * w34[f];
    atomicAdd(&pool[batch[i]], y);
}

__global__ void k_sigmoid(const float* __restrict__ pool, float* __restrict__ out, int G) {
    int g = blockIdx.x * blockDim.x + threadIdx.x;
    if (g < G) out[g] = 1.0f / (1.0f + __expf(-pool[g]));
}

// ---------------------------------------------------------------------------

static inline size_t align64(size_t x) { return (x + 63) & ~(size_t)63; }
static inline int    cdiv(long long a, long long b) { return (int)((a + b - 1) / b); }

extern "C" void kernel_launch(void* const* d_in, const int* in_sizes, int n_in,
                              void* d_out, int out_size, void* d_ws, size_t ws_size,
                              hipStream_t stream) {
    const float* x     = (const float*)d_in[0];
    const int*   edges = (const int*)  d_in[1];
    const int*   batch = (const int*)  d_in[2];
    const float* W1    = (const float*)d_in[3];
    const float* b1    = (const float*)d_in[4];
    const float* W2    = (const float*)d_in[5];
    const float* b2    = (const float*)d_in[6];
    const float* W3    = (const float*)d_in[7];
    const float* b3    = (const float*)d_in[8];
    const float* W4    = (const float*)d_in[9];
    const float* b4    = (const float*)d_in[10];
    float*       out   = (float*)d_out;

    const int Nn = in_sizes[2];        // 100000 nodes
    const int E  = in_sizes[1] / 2;    // 3.2M edges
    const int G  = out_size;           // 512 graphs
    const int* src = edges;
    const int* dst = edges + E;

    // ---- workspace layout (units: floats), ~59 MB -------------------------
    float* ws = (float*)d_ws;
    const size_t off_dis  = 0;                                  // dis [Nn] f32
    const size_t off_R1   = align64((size_t)Nn);                // Nn*72 floats
    const size_t off_R2   = off_R1 + align64((size_t)Nn * 72);  // Nn*72 floats
    const size_t off_bp1  = off_R2 + align64((size_t)Nn * 72);  // 12800 bf16
    const size_t off_bp2  = off_bp1 + align64(12800 / 2);       // 4608 bf16
    const size_t off_w34  = off_bp2 + align64(4608 / 2);
    const size_t off_pool = off_w34 + 64;

    float*  dis   = ws + off_dis;
    __bf16* h1    = (__bf16*)(ws + off_R1);  // [Nn,72] bf16
    __bf16* hr1   = (__bf16*)(ws + off_R1);  // relu'd copy, reuses h1 (h1 dead)
    float*  agg2f = (float*) (ws + off_R1);  // [Nn,40] f32 (hr1 dead after GEMM2)
    float*  agg1f = (float*) (ws + off_R2);  // [Nn,72] f32
    __bf16* h2    = (__bf16*)(ws + off_R2);  // [Nn,40] bf16 (agg1f dead after ReLU copy)
    __bf16* bp1   = (__bf16*)(ws + off_bp1);
    __bf16* bp2   = (__bf16*)(ws + off_bp2);
    float*  w34   = ws + off_w34;
    float*  pool  = ws + off_pool;

    const int T = 256;

    // degree with self-loops -> dis = deg^-1/2
    k_fill<<<cdiv(Nn, T), T, 0, stream>>>(dis, 1.0f, Nn);
    k_deg_count<<<cdiv(E, T), T, 0, stream>>>(dst, dis, E);
    k_rsqrt_inplace<<<cdiv(Nn, T), T, 0, stream>>>(dis, Nn);

    // prepack weights into bf16 WMMA B-fragments (one-off, tiny)
    k_pack_b<136, 68><<<cdiv(5 * 5 * 32, T), T, 0, stream>>>(W1, bp1);
    k_pack_b<68, 34><<<cdiv(3 * 3 * 32, T), T, 0, stream>>>(W2, bp2);

    // ---- layer 1: h1 = x @ W1 (K=136, N=68, stride 72, bf16 out) ----------
    {
        const int tiles = cdiv(Nn, 16) * 5;
        k_gemm_wmma<136, 68, 72, 136, float, __bf16>
            <<<cdiv(tiles, 4), 128, 0, stream>>>(x, bp1, h1, Nn);
    }
    k_selfloop_init<<<cdiv((long long)Nn * 72, T), T, 0, stream>>>(dis, h1, agg1f, Nn, 68, 72);
    k_edge_agg<<<cdiv((long long)E * 9, T), T, 0, stream>>>(src, dst, dis, h1, agg1f, E, 9, 72);
    k_bias_relu_bf16<<<cdiv((long long)Nn * 72, T), T, 0, stream>>>(agg1f, b1, hr1, Nn, 68, 72);

    // ---- layer 2: h2 = hr1 @ W2 (K=68, N=34, stride 40, bf16 in+out) ------
    {
        const int tiles = cdiv(Nn, 16) * 3;
        k_gemm_wmma<68, 34, 40, 72, __bf16, __bf16>
            <<<cdiv(tiles, 4), 128, 0, stream>>>(hr1, bp2, h2, Nn);
    }
    k_selfloop_init<<<cdiv((long long)Nn * 40, T), T, 0, stream>>>(dis, h2, agg2f, Nn, 34, 40);
    k_edge_agg<<<cdiv((long long)E * 5, T), T, 0, stream>>>(src, dst, dis, h2, agg2f, E, 5, 40);
    k_bias_relu_f32<<<cdiv((long long)Nn * 40, T), T, 0, stream>>>(agg2f, b2, Nn, 34, 40);

    // ---- fused tail: y = h2r @ (W3W4) + c, pooled per graph, sigmoid ------
    k_w34<<<1, 64, 0, stream>>>(W3, b3, W4, b4, w34);
    k_fill<<<cdiv(G, T), T, 0, stream>>>(pool, 0.0f, G);
    k_node_pool<<<cdiv(Nn, T), T, 0, stream>>>(agg2f, w34, batch, pool, Nn, 34, 40);
    k_sigmoid<<<cdiv(G, T), T, 0, stream>>>(pool, out, G);
}
